// BAE_14199161880953
// MI455X (gfx1250) — compile-verified, matrix operand order checked by reading, not run on hardware
//
#include <hip/hip_runtime.h>
#include <hip/hip_bf16.h>
#include <math.h>

typedef __attribute__((ext_vector_type(16))) _Float16 v16h;
typedef __attribute__((ext_vector_type(8)))  _Float16 v8h;
typedef __attribute__((ext_vector_type(8)))  float    v8f;
typedef __attribute__((ext_vector_type(4)))  int      v4i;

#define AS1 __attribute__((address_space(1)))
#define AS3 __attribute__((address_space(3)))

// gfx1250 async global->LDS copy path (ASYNCcnt-tracked). Guarded so the
// compile cannot break if the toolchain lacks the builtins.
// Probe-confirmed signature (from clang diagnostic): pointer params are
// address-space-qualified int4 vectors: (v4i AS1*, v4i AS3*, imm, imm).
#if defined(__gfx1250__) && \
    __has_builtin(__builtin_amdgcn_global_load_async_to_lds_b128) && \
    __has_builtin(__builtin_amdgcn_s_wait_asynccnt)
#define USE_ASYNC_LDS 1
#else
#define USE_ASYNC_LDS 0
#endif

#define EPI_BIAS    0
#define EPI_RELU_BN 1
#define EPI_TANH    2

// ---------------------------------------------------------------------------
// fp32 -> fp16 conversion with padded leading dimension (zero-filled pad).
// dst is [rows, Nld], src is [rows, N].
// ---------------------------------------------------------------------------
__global__ void cvt_f16_kernel(const float* __restrict__ src, _Float16* __restrict__ dst,
                               int rows, int N, int Nld)
{
    long idx = (long)blockIdx.x * blockDim.x + threadIdx.x;
    long total = (long)rows * Nld;
    if (idx >= total) return;
    int r = (int)(idx / Nld);
    int c = (int)(idx % Nld);
    dst[idx] = (c < N) ? (_Float16)src[(long)r * N + c] : (_Float16)0.0f;
}

// ---------------------------------------------------------------------------
// Tiled WMMA GEMM: epilogue(A[M,K]f16 @ W[K,Nw]f16 + bias) -> C32[M,N] / C16[M,N]
// v_wmma_f32_16x16x32_f16, fp32 accumulate. Block tile 128x64x32, 8 wave32
// waves (4x2), 2x2 WMMA tiles/wave. M % 128 == 0, K % 32 == 0, Nw % 8 == 0.
// A-tile staging uses GLOBAL_LOAD_ASYNC_TO_LDS_B128 when available.
// ---------------------------------------------------------------------------
template<int EPI, bool OUT32, bool OUT16>
__global__ __launch_bounds__(256)
void wmma_gemm_f16(const _Float16* __restrict__ A, const _Float16* __restrict__ W,
                   const float* __restrict__ bias,
                   const float* __restrict__ bn_g, const float* __restrict__ bn_b,
                   const float* __restrict__ bn_m, const float* __restrict__ bn_v,
                   float* __restrict__ C32, _Float16* __restrict__ C16,
                   int M, int N, int Nw, int K)
{
    constexpr int BM = 128, BN = 64, BK = 32;
    // +8 halves pad => row stride 80B (keeps 16B alignment, staggers banks)
    __shared__ __align__(16) _Float16 sA [BM][BK + 8];
    __shared__ __align__(16) _Float16 sBt[BN][BK + 8];   // B tile transposed [n][k]

    const int tid  = threadIdx.x;
    const int lane = tid & 31;
    const int wave = tid >> 5;
    const int wm   = wave >> 1;          // 0..3
    const int wn   = wave & 1;           // 0..1

    const int row0 = blockIdx.y * BM;
    const int col0 = blockIdx.x * BN;

    v8f acc[2][2] = {};

    // ISA 7.12.2: lanes 0-15 hold K {0..7,16..23}; lanes 16-31 hold K {8..15,24..31}
    const int khalf = (lane >> 4) * 8;
    const int mi    = lane & 15;

    for (int k0 = 0; k0 < K; k0 += BK) {
        // ---- stage A tile: 128x32 halves = 512 x 16B chunks, 2 per thread ----
        {
            const _Float16* Ab = A + (long)row0 * K + k0;
            #pragma unroll
            for (int it = 0; it < 2; ++it) {
                int id = tid + it * 256;          // 0..511
                int r  = id >> 2;                 // 0..127
                int c8 = (id & 3) << 3;           // 0,8,16,24
#if USE_ASYNC_LDS
                __builtin_amdgcn_global_load_async_to_lds_b128(
                    (AS1 v4i*)(Ab + (long)r * K + c8),
                    (AS3 v4i*)&sA[r][c8], 0, 0);
#else
                *(v8h*)&sA[r][c8] = *(const v8h*)(Ab + (long)r * K + c8);
#endif
            }
        }
        // ---- stage B tile transposed: 32x64, one 8-half chunk per thread ----
        {
            int r  = tid >> 3;                    // k 0..31
            int c8 = (tid & 7) << 3;              // n 0,8,...,56
            int n  = col0 + c8;
            v8h wv = {};
            if (n < Nw)                           // chunk-granular guard (Nw % 8 == 0)
                wv = *(const v8h*)(W + (long)(k0 + r) * Nw + n);
            #pragma unroll
            for (int e = 0; e < 8; ++e) sBt[c8 + e][r] = wv[e];
        }
#if USE_ASYNC_LDS
        __builtin_amdgcn_s_wait_asynccnt(0);
#endif
        __syncthreads();

        // speculative prefetch of next A K-slab -> global_prefetch_b8
        if (k0 + BK < K)
            __builtin_prefetch(A + (long)(row0 + (tid >> 1)) * K + (k0 + BK), 0, 1);

        // ---- fragments (contiguous 16B ds reads) + 2x2 WMMA ----
        v16h af[2], bf[2];
        #pragma unroll
        for (int i = 0; i < 2; ++i) {
            const _Float16* ap = &sA[wm * 32 + i * 16 + mi][khalf];
            v8h lo = *(const v8h*)ap;
            v8h hi = *(const v8h*)(ap + 16);
            af[i] = __builtin_shufflevector(lo, hi, 0,1,2,3,4,5,6,7,8,9,10,11,12,13,14,15);
        }
        #pragma unroll
        for (int j = 0; j < 2; ++j) {
            const _Float16* bp = &sBt[wn * 32 + j * 16 + mi][khalf];
            v8h lo = *(const v8h*)bp;
            v8h hi = *(const v8h*)(bp + 16);
            bf[j] = __builtin_shufflevector(lo, hi, 0,1,2,3,4,5,6,7,8,9,10,11,12,13,14,15);
        }
        #pragma unroll
        for (int i = 0; i < 2; ++i)
            #pragma unroll
            for (int j = 0; j < 2; ++j)
                acc[i][j] = __builtin_amdgcn_wmma_f32_16x16x32_f16(
                    false, af[i], false, bf[j], (short)0, acc[i][j], false, false);

        __syncthreads();
    }

    // ---- epilogue (C layout: lane n = lane&15, row = r + 8*(lane>>4)) ----
    const int nw   = lane & 15;
    const int moff = (lane >> 4) * 8;
    #pragma unroll
    for (int i = 0; i < 2; ++i) {
        #pragma unroll
        for (int j = 0; j < 2; ++j) {
            int col = col0 + wn * 32 + j * 16 + nw;
            if (col >= N) continue;
            float bv = bias ? bias[col] : 0.0f;
            float gg = 0.f, bb = 0.f, mm = 0.f, inv = 0.f;
            if (EPI == EPI_RELU_BN) {
                gg = bn_g[col]; bb = bn_b[col]; mm = bn_m[col];
                inv = rsqrtf(bn_v[col] + 1e-5f);
            }
            #pragma unroll
            for (int r = 0; r < 8; ++r) {
                int row = row0 + wm * 32 + i * 16 + moff + r;
                float v = acc[i][j][r] + bv;
                if (EPI == EPI_RELU_BN)   v = (fmaxf(v, 0.0f) - mm) * inv * gg + bb;
                else if (EPI == EPI_TANH) v = tanhf(v);
                if (OUT32) C32[(long)row * N + col] = v;
                if (OUT16) C16[(long)row * N + col] = (_Float16)v;
            }
        }
    }
}

// ---------------------------------------------------------------------------
// Combined double-softmax multipliers (optical x log on same indices):
//   node_mul[b][j] = softmax(on[b],10)[j] * softmax(ln[b],12)[j], j<10
//   edge_mul[b][e] = softmax(oe[b],55)[e] * softmax(le[b],78)[e], e<55
// ---------------------------------------------------------------------------
__global__ void softmax_combine_kernel(const float* __restrict__ on, const float* __restrict__ oe,
                                       const float* __restrict__ ln, const float* __restrict__ le,
                                       float* __restrict__ node_mul, float* __restrict__ edge_mul,
                                       int B)
{
    int b = blockIdx.x * blockDim.x + threadIdx.x;
    if (b >= B) return;
    {
        const float* po = on + (long)b * 10;
        const float* pl = ln + (long)b * 12;
        float mo = -1e30f, ml = -1e30f;
        for (int i = 0; i < 10; ++i) mo = fmaxf(mo, po[i]);
        for (int i = 0; i < 12; ++i) ml = fmaxf(ml, pl[i]);
        float so = 0.f, sl = 0.f;
        for (int i = 0; i < 10; ++i) so += __expf(po[i] - mo);
        for (int i = 0; i < 12; ++i) sl += __expf(pl[i] - ml);
        float io = 1.f / so, il = 1.f / sl;
        for (int i = 0; i < 10; ++i)
            node_mul[(long)b * 10 + i] = (__expf(po[i] - mo) * io) * (__expf(pl[i] - ml) * il);
    }
    {
        const float* po = oe + (long)b * 55;
        const float* pl = le + (long)b * 78;
        float mo = -1e30f, ml = -1e30f;
        for (int i = 0; i < 55; ++i) mo = fmaxf(mo, po[i]);
        for (int i = 0; i < 78; ++i) ml = fmaxf(ml, pl[i]);
        float so = 0.f, sl = 0.f;
        for (int i = 0; i < 55; ++i) so += __expf(po[i] - mo);
        for (int i = 0; i < 78; ++i) sl += __expf(pl[i] - ml);
        float io = 1.f / so, il = 1.f / sl;
        for (int i = 0; i < 55; ++i)
            edge_mul[(long)b * 55 + i] = (__expf(po[i] - mo) * io) * (__expf(pl[i] - ml) * il);
    }
}

// ---------------------------------------------------------------------------
// Apply attention multipliers, scatter to d_out (edge ++ node ++ mu ++ logvar).
// Edge index for (i,j), 20<=i<=j<=29:  e = (41-i)(i-20)/2 + (j-i)
// ---------------------------------------------------------------------------
__global__ void attn_scatter_kernel(const float* __restrict__ dec_out,
                                    const float* __restrict__ node_mul,
                                    const float* __restrict__ edge_mul,
                                    float* __restrict__ out, int B)
{
    long idx = (long)blockIdx.x * blockDim.x + threadIdx.x;
    long total = (long)B * 2820;
    if (idx >= total) return;
    int b = (int)(idx / 2820);
    int c = (int)(idx % 2820);
    float v = dec_out[idx];
    if (c < 900) {
        int i = c / 30, j = c % 30;
        if (i >= 20 && j >= i && j <= 29) {
            int e = ((41 - i) * (i - 20)) / 2 + (j - i);
            v *= edge_mul[(long)b * 55 + e];
        }
        out[(long)b * 900 + c] = v;
    } else {
        int c2 = c - 900;
        int i = c2 / 64;
        if (i >= 20 && i <= 29) v *= node_mul[(long)b * 10 + (i - 20)];
        out[(long)B * 900 + (long)b * 1920 + c2] = v;
    }
}

// ---------------------------------------------------------------------------
extern "C" void kernel_launch(void* const* d_in, const int* in_sizes, int n_in,
                              void* d_out, int out_size, void* d_ws, size_t ws_size,
                              hipStream_t stream)
{
    (void)in_sizes; (void)n_in; (void)out_size; (void)ws_size;
    const int B = 8192;

    const float* x     = (const float*)d_in[0];
    const float* opt   = (const float*)d_in[1];
    const float* lgi   = (const float*)d_in[2];
    const float* enc_W = (const float*)d_in[3];  const float* enc_b = (const float*)d_in[4];
    const float* bn1_g = (const float*)d_in[5];  const float* bn1_b = (const float*)d_in[6];
    const float* bn1_m = (const float*)d_in[7];  const float* bn1_v = (const float*)d_in[8];
    const float* mu_W  = (const float*)d_in[9];  const float* mu_b  = (const float*)d_in[10];
    const float* lv_W  = (const float*)d_in[11]; const float* lv_b  = (const float*)d_in[12];
    const float* dc_W1 = (const float*)d_in[13]; const float* dc_b1 = (const float*)d_in[14];
    const float* bn2_g = (const float*)d_in[15]; const float* bn2_b = (const float*)d_in[16];
    const float* bn2_m = (const float*)d_in[17]; const float* bn2_v = (const float*)d_in[18];
    const float* dc_W2 = (const float*)d_in[19]; const float* dc_b2 = (const float*)d_in[20];
    const float* onW1  = (const float*)d_in[21]; const float* onb1  = (const float*)d_in[22];
    const float* onW2  = (const float*)d_in[23]; const float* onb2  = (const float*)d_in[24];
    const float* oeW1  = (const float*)d_in[25]; const float* oeb1  = (const float*)d_in[26];
    const float* oeW2  = (const float*)d_in[27]; const float* oeb2  = (const float*)d_in[28];
    const float* lnW1  = (const float*)d_in[29]; const float* lnb1  = (const float*)d_in[30];
    const float* lnW2  = (const float*)d_in[31]; const float* lnb2  = (const float*)d_in[32];
    const float* leW1  = (const float*)d_in[33]; const float* leb1  = (const float*)d_in[34];
    const float* leW2  = (const float*)d_in[35]; const float* leb2  = (const float*)d_in[36];

    // d_out layout: edge[B*900] ++ node[B*1920] ++ mu[B*256] ++ logvar[B*256]
    float* out    = (float*)d_out;
    float* mu_out = out + (long)B * 2820;
    float* lv_out = mu_out + (long)B * 256;

    // ---- workspace bump allocator (256B aligned regions), ~220 MB ----
    char* wp = (char*)d_ws;
    auto alloc = [&](size_t bytes) -> void* {
        void* r = (void*)wp; wp += (bytes + 255) & ~(size_t)255; return r;
    };
    float*     dec_out  = (float*)alloc((size_t)B * 2820 * 4);
    _Float16*  h16      = (_Float16*)alloc((size_t)B * 2048 * 2);   // h1 then h2
    _Float16*  t16      = (_Float16*)alloc((size_t)B * 512 * 2);    // attn hidden (reused)
    _Float16*  mu16     = (_Float16*)alloc((size_t)B * 256 * 2);
    _Float16*  x16      = (_Float16*)alloc((size_t)B * 1024 * 2);
    _Float16*  opt16    = (_Float16*)alloc((size_t)B * 1024 * 2);
    _Float16*  lgi16    = (_Float16*)alloc((size_t)B * 1024 * 2);
    _Float16*  wenc     = (_Float16*)alloc((size_t)1024 * 2048 * 2);
    _Float16*  wmu      = (_Float16*)alloc((size_t)2048 * 256 * 2);
    _Float16*  wlv      = (_Float16*)alloc((size_t)2048 * 256 * 2);
    _Float16*  wdc1     = (_Float16*)alloc((size_t)256 * 2048 * 2);
    _Float16*  wdc2     = (_Float16*)alloc((size_t)2048 * 2824 * 2);  // Nld=2824
    _Float16*  won1     = (_Float16*)alloc((size_t)1024 * 512 * 2);
    _Float16*  woe1     = (_Float16*)alloc((size_t)1024 * 512 * 2);
    _Float16*  wln1     = (_Float16*)alloc((size_t)1024 * 512 * 2);
    _Float16*  wle1     = (_Float16*)alloc((size_t)1024 * 512 * 2);
    _Float16*  won2     = (_Float16*)alloc((size_t)512 * 16 * 2);     // Nld=16
    _Float16*  woe2     = (_Float16*)alloc((size_t)512 * 56 * 2);     // Nld=56
    _Float16*  wln2     = (_Float16*)alloc((size_t)512 * 16 * 2);     // Nld=16
    _Float16*  wle2     = (_Float16*)alloc((size_t)512 * 80 * 2);     // Nld=80
    float*     lg_on    = (float*)alloc((size_t)B * 10 * 4);
    float*     lg_oe    = (float*)alloc((size_t)B * 55 * 4);
    float*     lg_ln    = (float*)alloc((size_t)B * 12 * 4);
    float*     lg_le    = (float*)alloc((size_t)B * 78 * 4);
    float*     node_mul = (float*)alloc((size_t)B * 10 * 4);
    float*     edge_mul = (float*)alloc((size_t)B * 55 * 4);

    dim3 blk(256);
    auto cvt = [&](const float* s, _Float16* d, int rows, int N, int Nld) {
        long tot = (long)rows * Nld;
        cvt_f16_kernel<<<dim3((unsigned)((tot + 255) / 256)), blk, 0, stream>>>(s, d, rows, N, Nld);
    };
    auto grid = [](int M, int N) { return dim3((N + 63) / 64, (M + 127) / 128); };

    // ---- one-time fp32 -> fp16 shadows (padded ld for clean 16B tiles) ----
    cvt(x,     x16,   B, 1024, 1024);
    cvt(opt,   opt16, B, 1024, 1024);
    cvt(lgi,   lgi16, B, 1024, 1024);
    cvt(enc_W, wenc, 1024, 2048, 2048);
    cvt(mu_W,  wmu,  2048, 256, 256);
    cvt(lv_W,  wlv,  2048, 256, 256);
    cvt(dc_W1, wdc1, 256, 2048, 2048);
    cvt(dc_W2, wdc2, 2048, 2820, 2824);
    cvt(onW1,  won1, 1024, 512, 512);
    cvt(oeW1,  woe1, 1024, 512, 512);
    cvt(lnW1,  wln1, 1024, 512, 512);
    cvt(leW1,  wle1, 1024, 512, 512);
    cvt(onW2,  won2, 512, 10, 16);
    cvt(oeW2,  woe2, 512, 55, 56);
    cvt(lnW2,  wln2, 512, 12, 16);
    cvt(leW2,  wle2, 512, 78, 80);

    // ---- VAE path ----
    // h1 = BN(ReLU(x @ enc_W + b))            [f16 out only]
    wmma_gemm_f16<EPI_RELU_BN, false, true><<<grid(B, 2048), blk, 0, stream>>>(
        x16, wenc, enc_b, bn1_g, bn1_b, bn1_m, bn1_v, nullptr, h16, B, 2048, 2048, 1024);
    // mu -> d_out slice (f32) + f16 shadow for decoder input
    wmma_gemm_f16<EPI_BIAS, true, true><<<grid(B, 256), blk, 0, stream>>>(
        h16, wmu, mu_b, nullptr, nullptr, nullptr, nullptr, mu_out, mu16, B, 256, 256, 2048);
    // logvar -> d_out slice
    wmma_gemm_f16<EPI_BIAS, true, false><<<grid(B, 256), blk, 0, stream>>>(
        h16, wlv, lv_b, nullptr, nullptr, nullptr, nullptr, lv_out, nullptr, B, 256, 256, 2048);
    // h2 = BN(ReLU(mu @ dec_W1 + b))          [f16 out, reuses h16]
    wmma_gemm_f16<EPI_RELU_BN, false, true><<<grid(B, 2048), blk, 0, stream>>>(
        mu16, wdc1, dc_b1, bn2_g, bn2_b, bn2_m, bn2_v, nullptr, h16, B, 2048, 2048, 256);
    // dec_out = h2 @ dec_W2 + b               [dominant GEMM, N=2820, Nw=2824]
    wmma_gemm_f16<EPI_BIAS, true, false><<<grid(B, 2820), blk, 0, stream>>>(
        h16, wdc2, dc_b2, nullptr, nullptr, nullptr, nullptr, dec_out, nullptr, B, 2820, 2824, 2048);

    // ---- attention MLPs: logits = tanh(in @ W1 + b1) @ W2 + b2 ----
    wmma_gemm_f16<EPI_TANH, false, true><<<grid(B, 512), blk, 0, stream>>>(
        opt16, won1, onb1, nullptr, nullptr, nullptr, nullptr, nullptr, t16, B, 512, 512, 1024);
    wmma_gemm_f16<EPI_BIAS, true, false><<<grid(B, 10), blk, 0, stream>>>(
        t16, won2, onb2, nullptr, nullptr, nullptr, nullptr, lg_on, nullptr, B, 10, 16, 512);

    wmma_gemm_f16<EPI_TANH, false, true><<<grid(B, 512), blk, 0, stream>>>(
        opt16, woe1, oeb1, nullptr, nullptr, nullptr, nullptr, nullptr, t16, B, 512, 512, 1024);
    wmma_gemm_f16<EPI_BIAS, true, false><<<grid(B, 55), blk, 0, stream>>>(
        t16, woe2, oeb2, nullptr, nullptr, nullptr, nullptr, lg_oe, nullptr, B, 55, 56, 512);

    wmma_gemm_f16<EPI_TANH, false, true><<<grid(B, 512), blk, 0, stream>>>(
        lgi16, wln1, lnb1, nullptr, nullptr, nullptr, nullptr, nullptr, t16, B, 512, 512, 1024);
    wmma_gemm_f16<EPI_BIAS, true, false><<<grid(B, 12), blk, 0, stream>>>(
        t16, wln2, lnb2, nullptr, nullptr, nullptr, nullptr, lg_ln, nullptr, B, 12, 16, 512);

    wmma_gemm_f16<EPI_TANH, false, true><<<grid(B, 512), blk, 0, stream>>>(
        lgi16, wle1, leb1, nullptr, nullptr, nullptr, nullptr, nullptr, t16, B, 512, 512, 1024);
    wmma_gemm_f16<EPI_BIAS, true, false><<<grid(B, 78), blk, 0, stream>>>(
        t16, wle2, leb2, nullptr, nullptr, nullptr, nullptr, lg_le, nullptr, B, 78, 80, 512);

    // ---- double-softmax multipliers, then apply + scatter into d_out ----
    softmax_combine_kernel<<<dim3((B + 255) / 256), blk, 0, stream>>>(
        lg_on, lg_oe, lg_ln, lg_le, node_mul, edge_mul, B);

    long total = (long)B * 2820;
    attn_scatter_kernel<<<dim3((unsigned)((total + 255) / 256)), blk, 0, stream>>>(
        dec_out, node_mul, edge_mul, out, B);
}
